// model29_29145648071293
// MI455X (gfx1250) — matrix operand order repaired and checked by reading, not compiled, not run on hardware
//
#include <hip/hip_runtime.h>
#include <hip/hip_bf16.h>
#include <stdint.h>

// ---------------------------------------------------------------------------
// Model: per-graph GCN (29 nodes) -> fc1(29->128) -> fc2(128->128) -> fc(128->1296)
// Strategy: precompute normalized adjacency + f16 WMMA B-fragments once, then a
// single fused kernel: VALU GCN + three WMMA GEMM phases through LDS.
// Output stores are staged through per-wave LDS for 128B-coalesced NT stores.
// ---------------------------------------------------------------------------

typedef __attribute__((ext_vector_type(16))) _Float16 v16h;
typedef __attribute__((ext_vector_type(8)))  float    v8f;
typedef __attribute__((ext_vector_type(4)))  float    v4f;
typedef __attribute__((ext_vector_type(4)))  int      v4i;

#define N_NODES 29
#define M_BLK   64          // graphs per workgroup
#define THREADS 256         // 8 waves (wave32)

// ---- workspace layout (bytes) ----
#define WS_AHAT   0                      // f32 [29][32]          3712 B
#define WS_FRAG1  3712                   // i32 [8 frags][256]    8192 B   fc1 (K=32pad, N=128)
#define WS_FRAG2  11904                  // i32 [32 frags][256]  32768 B   fc2 (K=128, N=128)
#define WS_FRAGF  44672                  // i32 [324 frags][256] 331776 B  fc  (K=128, N=1296)

// ---- LDS layout (bytes), total 48256 ----
#define LDS_AHAT  0                      // f32 [29*32]   3712
#define LDS_H2    3712                   // f16 [64][32]  4096   (A of fc1, K padded to 32)
#define LDS_Y1    7808                   // f16 [64][128] 16384  (A of fc2; phase-4: store staging)
#define LDS_U     24192                  // union region: 24064
#define LDS_XW    (LDS_U)                //   f32 [64][61] 15616 (phase1 only)
#define LDS_HW    (LDS_U + 15616)        //   f32 [64][33]  8448 (phase1 only)
#define LDS_Y2    (LDS_U)                //   f16 [64][128] 16384 (A of fc)
#define SMEM_BYTES 48256

// ---------------------------------------------------------------------------
// prep kernel: build Ahat and repack weights into WMMA B-fragment layout.
// B-frag (32x16 f16) layout: lane n = lane&15; K = kt*32 + 2*(d&3) + (d>=4)*16 + (lane>=16)*8 + h
// stored lane-major: dst[f*256 + lane*8 + d] = pack(h0,h1)
// ---------------------------------------------------------------------------
__device__ __forceinline__ void build_frags(int* dst, const float* src,
                                            int Kreal, int N, int KT, int NT,
                                            int tid, int nthreads) {
  const int total = KT * NT * 256;
  for (int idx = tid; idx < total; idx += nthreads) {
    const int f    = idx >> 8;
    const int r    = idx & 255;
    const int lane = r >> 3;
    const int d    = r & 7;
    const int kt   = f % KT;
    const int nt   = f / KT;
    const int hi   = (lane >> 4) & 1;
    const int n    = nt * 16 + (lane & 15);
    const int k0   = kt * 32 + 2 * (d & 3) + ((d >> 2) << 4) + hi * 8;
    const float v0 = (k0     < Kreal) ? src[(size_t)k0 * N + n]       : 0.0f;
    const float v1 = (k0 + 1 < Kreal) ? src[(size_t)(k0 + 1) * N + n] : 0.0f;
    union { _Float16 h[2]; int i; } P;
    P.h[0] = (_Float16)v0;
    P.h[1] = (_Float16)v1;
    dst[idx] = P.i;
  }
}

__global__ void gcn_prep_kernel(const int* __restrict__ ei, int E,
                                const float* __restrict__ fc1W,
                                const float* __restrict__ fc2W,
                                const float* __restrict__ fcW,
                                float* __restrict__ Ahat,
                                int* __restrict__ frag1,
                                int* __restrict__ frag2,
                                int* __restrict__ fragF) {
  const int tid = blockIdx.x * blockDim.x + threadIdx.x;
  const int nthreads = gridDim.x * blockDim.x;

  if (tid == 0) {
    // Ahat[c][r] = dinv[r]*dinv[c] per edge (r->c), + dinv[i]^2 self loops.
    float deg[N_NODES], dinv[N_NODES];
    for (int i = 0; i < N_NODES; ++i) deg[i] = 1.0f;  // self loops
    const int* row = ei;
    const int* col = ei + E;
    for (int e = 0; e < E; ++e) deg[col[e]] += 1.0f;
    for (int i = 0; i < N_NODES; ++i) dinv[i] = rsqrtf(deg[i]);
    for (int i = 0; i < N_NODES; ++i)
      for (int j = 0; j < 32; ++j) Ahat[i * 32 + j] = 0.0f;
    for (int e = 0; e < E; ++e) {
      const int r = row[e], c = col[e];
      Ahat[c * 32 + r] += dinv[r] * dinv[c];
    }
    for (int i = 0; i < N_NODES; ++i) Ahat[i * 32 + i] += dinv[i] * dinv[i];
  }

  build_frags(frag1, fc1W, N_NODES, 128, 1, 8,  tid, nthreads);  // K=29 -> pad 32
  build_frags(frag2, fc2W, 128,     128, 4, 8,  tid, nthreads);
  build_frags(fragF, fcW,  128,    1296, 4, 81, tid, nthreads);
}

// ---------------------------------------------------------------------------
// Fragment loaders
// ---------------------------------------------------------------------------
// A-frag (16x32 f16) from row-major f16 LDS. ISA 7.12.2: lane m=lane&15;
// lanes 0-15: V0..3 -> K=0..7, V4..7 -> K=16..23; lanes 16-31: +8.
__device__ __forceinline__ v16h load_a_frag(const _Float16* base, int strideH,
                                            int row0, int kBase, int lane) {
  const int m  = lane & 15;
  const int hi = (lane >> 4) & 1;
  const unsigned* rp = (const unsigned*)(base + (row0 + m) * strideH + kBase);
  union { unsigned u[8]; v16h h; } U;
#pragma unroll
  for (int v = 0; v < 8; ++v) {
    const int k0 = 2 * v + ((v & 4) << 1) + (hi << 3);  // 2v + (v>=4?8:0) + hi*8
    U.u[v] = rp[k0 >> 1];
  }
  return U.h;
}

// B-frag: prepacked lane-major, two coalesced b128 loads per lane.
__device__ __forceinline__ v16h load_b_frag(const int* fragbase, int f, int lane) {
  union { v4i q[2]; v16h h; } U;
  const v4i* p = (const v4i*)(fragbase + f * 256 + lane * 8);
  U.q[0] = p[0];
  U.q[1] = p[1];
  return U.h;
}

__device__ __forceinline__ v8f wmma_f16(v16h a, v16h b, v8f c) {
  return __builtin_amdgcn_wmma_f32_16x16x32_f16(false, a, false, b,
                                                (short)0, c, false, false);
}

// ---------------------------------------------------------------------------
// Fused main kernel: 64 graphs per block, 8 waves.
// ---------------------------------------------------------------------------
__global__ void __launch_bounds__(THREADS)
gcn_main_kernel(const float* __restrict__ feat,
                const float* __restrict__ W1, const float* __restrict__ b1,
                const float* __restrict__ W2, const float* __restrict__ b2,
                const float* __restrict__ fc1b,
                const float* __restrict__ fc2b,
                const float* __restrict__ fcb,
                const float* __restrict__ AhatG,
                const int* __restrict__ frag1,
                const int* __restrict__ frag2,
                const int* __restrict__ fragF,
                float* __restrict__ out, int Btot) {
  extern __shared__ char smem[];
  float*    sA  = (float*)(smem + LDS_AHAT);
  _Float16* sH2 = (_Float16*)(smem + LDS_H2);
  _Float16* sY1 = (_Float16*)(smem + LDS_Y1);
  _Float16* sY2 = (_Float16*)(smem + LDS_Y2);
  float*    sXW = (float*)(smem + LDS_XW);
  float*    sHW = (float*)(smem + LDS_HW);

  const int tid   = threadIdx.x;
  const int lane  = tid & 31;
  const int wave  = tid >> 5;          // 0..7
  const int gbase = blockIdx.x * M_BLK;

  // Ahat -> LDS
  for (int i = tid; i < N_NODES * 32; i += THREADS) sA[i] = AhatG[i];

  // ---------------- Phase 1: per-graph GCN (4 threads / graph) --------------
  {
    const int gl  = tid >> 2;          // 0..63 local graph
    const int nst = tid & 3;
    const int g   = gbase + gl;
    const bool act = (g < Btot);
    const float w10 = W1[0], w11 = W1[1], w12 = W1[2],
                w13 = W1[3], w14 = W1[4], w15 = W1[5];
    const float bb0 = b1[0], bb1 = b1[1];
    const float w20 = W2[0], w21 = W2[1], bb2 = b2[0];
    float* xw = sXW + gl * 61;
    float* hw = sHW + gl * 33;
    _Float16* h2 = sH2 + gl * 32;

    if (act) {
      const float* xp = feat + (size_t)g * (N_NODES * 3);
      for (int n = nst; n < N_NODES; n += 4) {
        // features are streamed exactly once: non-temporal loads
        const float x0 = __builtin_nontemporal_load(xp + n * 3);
        const float x1 = __builtin_nontemporal_load(xp + n * 3 + 1);
        const float x2 = __builtin_nontemporal_load(xp + n * 3 + 2);
        xw[2 * n]     = x0 * w10 + x1 * w12 + x2 * w14;
        xw[2 * n + 1] = x0 * w11 + x1 * w13 + x2 * w15;
      }
    }
    __syncthreads();   // Ahat copy + xw visible
    if (act) {
      for (int n = nst; n < N_NODES; n += 4) {
        float a0 = bb0, a1 = bb1;
        for (int m = 0; m < N_NODES; ++m) {
          const float a = sA[n * 32 + m];
          a0 += a * xw[2 * m];
          a1 += a * xw[2 * m + 1];
        }
        const float h10 = fmaxf(a0, 0.0f), h11 = fmaxf(a1, 0.0f);
        hw[n] = h10 * w20 + h11 * w21;
      }
    }
    __syncthreads();
    if (act) {
      for (int n = nst; n < N_NODES; n += 4) {
        float a = bb2;
        for (int m = 0; m < N_NODES; ++m) a += sA[n * 32 + m] * hw[m];
        h2[n] = (_Float16)fmaxf(a, 0.0f);
      }
    } else {
      for (int n = nst; n < N_NODES; n += 4) h2[n] = (_Float16)0.0f;
    }
    if (nst == 0) { h2[29] = (_Float16)0.0f; h2[30] = (_Float16)0.0f; h2[31] = (_Float16)0.0f; }
  }
  __syncthreads();

  const int mt = wave & 3;             // 4 row tiles of 16 graphs
  const int wq = wave >> 2;            // column-half of the wave grid
  const int nn = lane & 15;
  const int hi = (lane >> 4) & 1;

  // ---------------- Phase 2: fc1  [64,32] x [32,128] -> relu -> sY1 ---------
  {
    const v16h aF = load_a_frag(sH2, 32, mt * 16, 0, lane);
#pragma unroll
    for (int j = 0; j < 4; ++j) {
      const int nt = wq * 4 + j;
      const v16h bF = load_b_frag(frag1, nt, lane);
      v8f acc = {0.f, 0.f, 0.f, 0.f, 0.f, 0.f, 0.f, 0.f};
      acc = wmma_f16(aF, bF, acc);
      const int   col  = nt * 16 + nn;
      const float bias = fc1b[col];
#pragma unroll
      for (int r = 0; r < 8; ++r) {
        const int row = mt * 16 + r + hi * 8;
        sY1[row * 128 + col] = (_Float16)fmaxf(acc[r] + bias, 0.0f);
      }
    }
  }
  __syncthreads();

  // ---------------- Phase 3: fc2  [64,128] x [128,128] -> relu -> sY2 -------
  {
    v16h aF[4];
#pragma unroll
    for (int kt = 0; kt < 4; ++kt)
      aF[kt] = load_a_frag(sY1, 128, mt * 16, kt * 32, lane);
#pragma unroll
    for (int j = 0; j < 4; ++j) {
      const int nt = wq * 4 + j;
      v8f acc = {0.f, 0.f, 0.f, 0.f, 0.f, 0.f, 0.f, 0.f};
#pragma unroll
      for (int kt = 0; kt < 4; ++kt) {
        const v16h bF = load_b_frag(frag2, nt * 4 + kt, lane);
        acc = wmma_f16(aF[kt], bF, acc);
      }
      const int   col  = nt * 16 + nn;
      const float bias = fc2b[col];
#pragma unroll
      for (int r = 0; r < 8; ++r) {
        const int row = mt * 16 + r + hi * 8;
        sY2[row * 128 + col] = (_Float16)fmaxf(acc[r] + bias, 0.0f);
      }
    }
  }
  __syncthreads();

  // ---------------- Phase 4: fc  [64,128] x [128,1296] + bias -> out --------
  // N-tiles processed in pairs; results staged via per-wave LDS scratch
  // (reusing dead sY1) so stores go out as 128B-per-row coalesced NT b128s.
  {
    v16h aF[4];
#pragma unroll
    for (int kt = 0; kt < 4; ++kt)
      aF[kt] = load_a_frag(sY2, 128, mt * 16, kt * 32, lane);

    float* scratch = (float*)(smem + LDS_Y1) + wave * 512;  // 16 rows x 32 cols

    const int pBeg = wq ? 20 : 0;          // pair index: nt = 2p, 2p+1
    const int pEnd = wq ? 40 : 20;
    for (int p = pBeg; p < pEnd; ++p) {
      const int nt0 = p * 2;
      if (p + 1 < pEnd)  // prefetch next pair's fragments (global_prefetch_b8)
        __builtin_prefetch(fragF + (nt0 + 2) * 4 * 256 + lane * 8, 0, 0);

      v8f acc0 = {0.f, 0.f, 0.f, 0.f, 0.f, 0.f, 0.f, 0.f};
      v8f acc1 = {0.f, 0.f, 0.f, 0.f, 0.f, 0.f, 0.f, 0.f};
#pragma unroll
      for (int kt = 0; kt < 4; ++kt) {
        const v16h b0 = load_b_frag(fragF, nt0 * 4 + kt, lane);
        const v16h b1 = load_b_frag(fragF, (nt0 + 1) * 4 + kt, lane);
        acc0 = wmma_f16(aF[kt], b0, acc0);
        acc1 = wmma_f16(aF[kt], b1, acc1);
      }
      const float bias0 = fcb[nt0 * 16 + nn];
      const float bias1 = fcb[nt0 * 16 + 16 + nn];
#pragma unroll
      for (int r = 0; r < 8; ++r) {
        const int rowl = r + hi * 8;
        scratch[rowl * 32 + nn]      = acc0[r] + bias0;
        scratch[rowl * 32 + 16 + nn] = acc1[r] + bias1;
      }
      // wave-private LDS round trip: DS ops from one wave are in-order.
#pragma unroll
      for (int it = 0; it < 4; ++it) {
        const int f    = it * 32 + lane;   // float4 index in 16x32 slab
        const int rowl = f >> 3;
        const int c4   = f & 7;
        const int rowg = gbase + mt * 16 + rowl;
        const v4f val  = *(const v4f*)(scratch + f * 4);
        if (rowg < Btot)
          __builtin_nontemporal_store(
              val, (v4f*)(out + (size_t)rowg * 1296 + nt0 * 16 + c4 * 4));
      }
    }

    if (wq == 1) {  // leftover tile nt = 80 (81 is odd)
      const int nt = 80;
      v8f acc = {0.f, 0.f, 0.f, 0.f, 0.f, 0.f, 0.f, 0.f};
#pragma unroll
      for (int kt = 0; kt < 4; ++kt) {
        const v16h bF = load_b_frag(fragF, nt * 4 + kt, lane);
        acc = wmma_f16(aF[kt], bF, acc);
      }
      const int   col  = nt * 16 + nn;
      const float bias = fcb[col];
#pragma unroll
      for (int r = 0; r < 8; ++r) {
        const int rowg = gbase + mt * 16 + r + hi * 8;
        if (rowg < Btot)
          __builtin_nontemporal_store(acc[r] + bias,
                                      out + (size_t)rowg * 1296 + col);
      }
    }
  }
}

// ---------------------------------------------------------------------------
extern "C" void kernel_launch(void* const* d_in, const int* in_sizes, int n_in,
                              void* d_out, int out_size, void* d_ws, size_t ws_size,
                              hipStream_t stream) {
  const float* feat = (const float*)d_in[0];
  const int*   ei   = (const int*)d_in[1];
  const float* W1   = (const float*)d_in[2];
  const float* b1   = (const float*)d_in[3];
  const float* W2   = (const float*)d_in[4];
  const float* b2   = (const float*)d_in[5];
  const float* fc1W = (const float*)d_in[6];
  const float* fc1b = (const float*)d_in[7];
  const float* fc2W = (const float*)d_in[8];
  const float* fc2b = (const float*)d_in[9];
  const float* fcW  = (const float*)d_in[10];
  const float* fcb  = (const float*)d_in[11];
  float* out = (float*)d_out;

  const int E    = in_sizes[1] / 2;                 // 116
  const int Btot = in_sizes[0] / (N_NODES * 3);     // 131072

  float* Ahat  = (float*)((char*)d_ws + WS_AHAT);
  int*   frag1 = (int*)((char*)d_ws + WS_FRAG1);
  int*   frag2 = (int*)((char*)d_ws + WS_FRAG2);
  int*   fragF = (int*)((char*)d_ws + WS_FRAGF);

  gcn_prep_kernel<<<128, THREADS, 0, stream>>>(ei, E, fc1W, fc2W, fcW,
                                               Ahat, frag1, frag2, fragF);

  const int blocks = (Btot + M_BLK - 1) / M_BLK;    // 2048
  gcn_main_kernel<<<blocks, THREADS, SMEM_BYTES, stream>>>(
      feat, W1, b1, W2, b2, fc1b, fc2b, fcb,
      Ahat, frag1, frag2, fragF, out, Btot);
}
